// Model_41721312313808
// MI455X (gfx1250) — matrix operand order repaired and checked by reading, not compiled, not run on hardware
//
#include <hip/hip_runtime.h>
#include <hip/hip_bf16.h>

// Model dims (fixed by the reference)
#define B_ 8
#define L_ 1024
#define DV_ 32
#define EH_ 256
#define DM_ 512
#define DF_ 1024
#define NS_ 16
#define BL_ (B_ * L_)   // 8192

typedef __attribute__((ext_vector_type(16))) __bf16 v16bf;
typedef __attribute__((ext_vector_type(8)))  float  v8f;

// ---------------------------------------------------------------------------
// 1) Per-(b,dv) moments over L for max_idx / min_idx (ddof=1 std).
// ---------------------------------------------------------------------------
__global__ __launch_bounds__(256) void stats_kernel(
    const float* __restrict__ maxIdx, const float* __restrict__ minIdx,
    float* meanMax, float* stdMax, float* meanMin, float* stdMin) {
  int bd = blockIdx.x;              // b*DV + dv
  int b = bd / DV_, dv = bd % DV_;
  const float* pMax = maxIdx + (size_t)b * L_ * DV_ + dv;
  const float* pMin = minIdx + (size_t)b * L_ * DV_ + dv;
  float s1 = 0.f, q1 = 0.f, s2 = 0.f, q2 = 0.f;
  for (int l = threadIdx.x; l < L_; l += 256) {
    float a = pMax[(size_t)l * DV_]; s1 += a; q1 += a * a;
    float c = pMin[(size_t)l * DV_]; s2 += c; q2 += c * c;
  }
  __shared__ float sh[4][256];
  sh[0][threadIdx.x] = s1; sh[1][threadIdx.x] = q1;
  sh[2][threadIdx.x] = s2; sh[3][threadIdx.x] = q2;
  __syncthreads();
  for (int off = 128; off > 0; off >>= 1) {
    if ((int)threadIdx.x < off)
      for (int i = 0; i < 4; ++i) sh[i][threadIdx.x] += sh[i][threadIdx.x + off];
    __syncthreads();
  }
  if (threadIdx.x == 0) {
    float m1 = sh[0][0] / L_, m2 = sh[2][0] / L_;
    float v1 = (sh[1][0] - (float)L_ * m1 * m1) / (float)(L_ - 1);
    float v2 = (sh[3][0] - (float)L_ * m2 * m2) / (float)(L_ - 1);
    meanMax[bd] = m1; stdMax[bd] = sqrtf(fmaxf(v1, 0.f));
    meanMin[bd] = m2; stdMin[bd] = sqrtf(fmaxf(v2, 0.f));
  }
}

// ---------------------------------------------------------------------------
// 2) Gating: mx/mn rows in LDS, 32x32 decay matmuls, softmax-of-2 blend -> z
// ---------------------------------------------------------------------------
__global__ __launch_bounds__(256) void gate_kernel(
    const float* __restrict__ seqx, const float* __restrict__ mask,
    const float* __restrict__ maxIdx, const float* __restrict__ minIdx,
    const float* __restrict__ maxVal, const float* __restrict__ minVal,
    const float* __restrict__ meanMax, const float* __restrict__ stdMax,
    const float* __restrict__ meanMin, const float* __restrict__ stdMin,
    const float* __restrict__ d1w, const float* __restrict__ d1b,
    const float* __restrict__ d2w, const float* __restrict__ d2b,
    float* __restrict__ z) {
  int ri = threadIdx.x >> 5, j = threadIdx.x & 31;
  int row = blockIdx.x * 8 + ri;    // 0..BL-1
  int b = row / L_;
  size_t idx = (size_t)row * DV_ + j;
  __shared__ float smx[8][32], smn[8][32];
  float mxv = (maxIdx[idx] - meanMax[b * DV_ + j]) / stdMax[b * DV_ + j];
  float mnv = (mxv - meanMin[b * DV_ + j]) / stdMin[b * DV_ + j];
  smx[ri][j] = mxv; smn[ri][j] = mnv;
  __syncthreads();
  float a1 = d1b[j], a2 = d2b[j];
  #pragma unroll 8
  for (int t = 0; t < 32; ++t) {
    a1 += smx[ri][t] * d1w[j * 32 + t];
    a2 += smn[ri][t] * d2w[j * 32 + t];
  }
  float g2 = expf(-a1), g1 = expf(-a2);
  float gs = g1 + g2; g1 /= gs; g2 /= gs;
  float mk = mask[idx];
  z[idx] = (g1 * maxVal[idx] + g2 * minVal[idx]) * (1.f - mk) + seqx[idx] * mk;
}

// ---------------------------------------------------------------------------
// 3) k=3 same-pad conv over L (32ch -> 256ch, relu) + mask projection.
// ---------------------------------------------------------------------------
__global__ __launch_bounds__(256) void conv_mask_kernel(
    const float* __restrict__ z, const float* __restrict__ mask,
    const float* __restrict__ mw, const float* __restrict__ mb,
    const float* __restrict__ maskw, const float* __restrict__ maskb,
    float* __restrict__ cbuf, float* __restrict__ ubuf) {
  int row = blockIdx.x;             // b*L + l
  int b = row / L_, l = row % L_;
  int e = threadIdx.x;              // 0..255
  __shared__ float zr[96];          // [k][dv]
  __shared__ float mr[32];
  if (e < 96) {
    int k = e >> 5, dv = e & 31;
    int ls = l + k - 1;
    zr[e] = (ls >= 0 && ls < L_) ? z[((size_t)b * L_ + ls) * DV_ + dv] : 0.f;
  }
  if (e < 32) mr[e] = mask[(size_t)row * DV_ + e];
  __syncthreads();
  const float* w = mw + (size_t)e * 96;   // mem_w[e][dv][k]
  float acc = mb[e];
  #pragma unroll
  for (int dv = 0; dv < 32; ++dv)
    acc += zr[dv] * w[dv * 3 + 0] + zr[32 + dv] * w[dv * 3 + 1] + zr[64 + dv] * w[dv * 3 + 2];
  float c = fmaxf(acc, 0.f);
  float m = maskb[e];
  #pragma unroll
  for (int dv = 0; dv < 32; ++dv) m += mr[dv] * maskw[e * 32 + dv];
  size_t oi = (size_t)row * EH_ + e;
  cbuf[oi] = c;
  ubuf[oi] = c * m;
}

// ---------------------------------------------------------------------------
// 4) Diagonal SSM scan: one thread per (b,h), 16 states in registers,
//    serial over L, coalesced over h. Software-pipelined: next step's u/x
//    loads are issued before the current state update so a global-load
//    latency overlaps the 32-FMA recurrence. Xout = Xin + y.
// ---------------------------------------------------------------------------
__global__ __launch_bounds__(256) void ssm_scan_kernel(
    const float* U, const float* Xin,
    const float* __restrict__ loglam, const float* __restrict__ Bm,
    const float* __restrict__ Cm, const float* __restrict__ Dp,
    float* __restrict__ Xout, int H) {
  int t = blockIdx.x * 256 + threadIdx.x;
  if (t >= B_ * H) return;
  int b = t / H, h = t % H;
  float lam[NS_], Bv[NS_], Cv[NS_], s[NS_];
  #pragma unroll
  for (int n = 0; n < NS_; ++n) {
    lam[n] = expf(-expf(loglam[h * NS_ + n]));
    Bv[n]  = Bm[h * NS_ + n];
    Cv[n]  = Cm[h * NS_ + n];
    s[n]   = 0.f;
  }
  float Dh = Dp[h];
  const float* up = U    + (size_t)b * L_ * H + h;
  const float* xp = Xin  + (size_t)b * L_ * H + h;
  float*       op = Xout + (size_t)b * L_ * H + h;
  float u_cur = up[0];
  float x_cur = xp[0];
  for (int l = 0; l < L_; ++l) {
    float u_nxt = 0.f, x_nxt = 0.f;
    if (l + 1 < L_) {                       // prefetch next step
      u_nxt = up[(size_t)(l + 1) * H];
      x_nxt = xp[(size_t)(l + 1) * H];
    }
    float u = u_cur;
    float y = Dh * u;
    #pragma unroll
    for (int n = 0; n < NS_; ++n) {
      s[n] = lam[n] * s[n] + Bv[n] * u;
      y += s[n] * Cv[n];
    }
    op[(size_t)l * H] = x_cur + y;
    u_cur = u_nxt; x_cur = x_nxt;
  }
}

// ---------------------------------------------------------------------------
// 5) LayerNorm over H per row (population variance, eps=1e-5).
// ---------------------------------------------------------------------------
__global__ __launch_bounds__(256) void ln_kernel(
    const float* __restrict__ X, const float* __restrict__ g,
    const float* __restrict__ bta, float* __restrict__ Y, int H) {
  int row = blockIdx.x;
  const float* x = X + (size_t)row * H;
  float* y = Y + (size_t)row * H;
  float s = 0.f, q = 0.f;
  for (int i = threadIdx.x; i < H; i += 256) { float v = x[i]; s += v; q += v * v; }
  __shared__ float sh[2][256];
  sh[0][threadIdx.x] = s; sh[1][threadIdx.x] = q;
  __syncthreads();
  for (int off = 128; off > 0; off >>= 1) {
    if ((int)threadIdx.x < off) {
      sh[0][threadIdx.x] += sh[0][threadIdx.x + off];
      sh[1][threadIdx.x] += sh[1][threadIdx.x + off];
    }
    __syncthreads();
  }
  float mu  = sh[0][0] / H;
  float var = sh[1][0] / H - mu * mu;
  float r = rsqrtf(var + 1e-5f);
  for (int i = threadIdx.x; i < H; i += 256)
    y[i] = (x[i] - mu) * r * g[i] + bta[i];
}

// ---------------------------------------------------------------------------
// 6) f32 -> bf16 weight conversion (once per GEMM, per launch).
// ---------------------------------------------------------------------------
__global__ __launch_bounds__(256) void cvt_bf16_kernel(
    const float* __restrict__ src, __bf16* __restrict__ dst, int n) {
  int i = blockIdx.x * 256 + threadIdx.x;
  if (i < n) dst[i] = (__bf16)src[i];
}

// ---------------------------------------------------------------------------
// 7) GEMM  C[M,N] = A[M,K] * W[N,K]^T (+bias, optional relu, optional resid)
//    bf16 WMMA 16x16x32, f32 accumulate. Wave tile = 16(M) x (16*NT)(N);
//    8 waves stacked in M => block tile 128 x 16*NT. The A fragment (and its
//    f32->bf16 conversion) is amortized over NT WMMAs per K-step.
//    M%128==0, N%(16*NT)==0, K%32==0 for all call sites.
//    Fragment layouts per CDNA5 ISA 7.12.2:
//      A: lane r=lane&15 = row, two 8-elem K chunks at kb+8h, kb+16+8h (h=lane>>4)
//      B: lane r = column (row r of W[N,K]), two 8-elem K chunks at kb+8h, kb+16+8h
//      D: acc[i] -> (m = i + 8h, n = lane&15)
// ---------------------------------------------------------------------------
template <int NT>
__global__ __launch_bounds__(256) void gemm_bf16_wmma(
    const float* __restrict__ A, const __bf16* __restrict__ W,
    const float* __restrict__ bias, const float* Res,
    float* __restrict__ C, int M, int N, int K, int doRelu) {
  int wave = threadIdx.x >> 5;
  int lane = threadIdx.x & 31;
  int r = lane & 15;
  int h = lane >> 4;
  int mBase = blockIdx.x * 128 + wave * 16;
  int nBase = blockIdx.y * (16 * NT);
  v8f acc[NT];
  #pragma unroll
  for (int t = 0; t < NT; ++t) acc[t] = (v8f){};
  const float* Arow = A + (size_t)(mBase + r) * K;
  const __bf16* Wrow[NT];
  #pragma unroll
  for (int t = 0; t < NT; ++t) Wrow[t] = W + (size_t)(nBase + 16 * t + r) * K;
  for (int kb = 0; kb < K; kb += 32) {
    v16bf a;
    #pragma unroll
    for (int e = 0; e < 8; ++e) {
      a[e]     = (__bf16)Arow[kb + 8 * h + e];
      a[e + 8] = (__bf16)Arow[kb + 16 + 8 * h + e];
    }
    v16bf bfrag[NT];
    #pragma unroll
    for (int t = 0; t < NT; ++t)
      #pragma unroll
      for (int e = 0; e < 16; ++e) bfrag[t][e] = Wrow[t][kb + 8 * h + e + (e >= 8 ? 8 : 0)];
    #pragma unroll
    for (int t = 0; t < NT; ++t)
      acc[t] = __builtin_amdgcn_wmma_f32_16x16x32_bf16(
          false, a, false, bfrag[t], (short)0, acc[t], false, false);
  }
  #pragma unroll
  for (int t = 0; t < NT; ++t) {
    int n = nBase + 16 * t + r;
    float bi = bias[n];
    #pragma unroll
    for (int i = 0; i < 8; ++i) {
      int m = mBase + i + 8 * h;
      size_t o = (size_t)m * N + n;
      float v = acc[t][i] + bi;
      if (doRelu) v = fmaxf(v, 0.f);
      if (Res)    v += Res[o];
      C[o] = v;
    }
  }
}

// ---------------------------------------------------------------------------
// Host side
// ---------------------------------------------------------------------------
struct S4Params {
  const float *loglam, *Bm, *Cm, *D, *n1g, *n1b, *n2g, *n2b, *c1w, *c1b, *c2w, *c2b;
};

static void launch_gemm(hipStream_t s, const float* A, const __bf16* W,
                        const float* bias, const float* Res, float* C,
                        int M, int N, int K, int relu) {
  if (N % 64 == 0) {
    dim3 grid(M / 128, N / 64);
    gemm_bf16_wmma<4><<<grid, 256, 0, s>>>(A, W, bias, Res, C, M, N, K, relu);
  } else {
    dim3 grid(M / 128, N / 32);
    gemm_bf16_wmma<2><<<grid, 256, 0, s>>>(A, W, bias, Res, C, M, N, K, relu);
  }
}

static void run_s4(hipStream_t s, const S4Params& P, const float* U, const float* Xin,
                   int H, float* xa, float* xln, float* ffh, float* y2,
                   __bf16* wbf, float* out) {
  ssm_scan_kernel<<<(B_ * H + 255) / 256, 256, 0, s>>>(U, Xin, P.loglam, P.Bm, P.Cm, P.D, xa, H);
  ln_kernel<<<BL_, 256, 0, s>>>(xa, P.n1g, P.n1b, xln, H);
  int n1 = DF_ * H;
  cvt_bf16_kernel<<<(n1 + 255) / 256, 256, 0, s>>>(P.c1w, wbf, n1);
  launch_gemm(s, xln, wbf, P.c1b, nullptr, ffh, BL_, DF_, H, 1);
  int n2 = H * DF_;
  cvt_bf16_kernel<<<(n2 + 255) / 256, 256, 0, s>>>(P.c2w, wbf, n2);
  launch_gemm(s, ffh, wbf, P.c2b, xln, y2, BL_, H, DF_, 0);
  ln_kernel<<<BL_, 256, 0, s>>>(y2, P.n2g, P.n2b, out, H);
}

extern "C" void kernel_launch(void* const* d_in, const int* in_sizes, int n_in,
                              void* d_out, int out_size, void* d_ws, size_t ws_size,
                              hipStream_t stream) {
  (void)in_sizes; (void)n_in; (void)out_size; (void)ws_size;
  // setup_inputs() dict order, params dict/list expanded in insertion order.
  const float* seq_x   = (const float*)d_in[0];
  const float* mask    = (const float*)d_in[1];
  const float* max_idx = (const float*)d_in[2];
  const float* min_idx = (const float*)d_in[3];
  const float* max_val = (const float*)d_in[4];
  const float* min_val = (const float*)d_in[5];
  int p = 6;
  const float* d1w = (const float*)d_in[p++]; const float* d1b = (const float*)d_in[p++];
  const float* d2w = (const float*)d_in[p++]; const float* d2b = (const float*)d_in[p++];
  const float* mem_w = (const float*)d_in[p++]; const float* mem_b = (const float*)d_in[p++];
  const float* mask_w = (const float*)d_in[p++]; const float* mask_b = (const float*)d_in[p++];
  const float* lin1_w = (const float*)d_in[p++]; const float* lin1_b = (const float*)d_in[p++];
  const float* norm_g = (const float*)d_in[p++]; const float* norm_b = (const float*)d_in[p++];
  const float* out_w = (const float*)d_in[p++]; const float* out_b = (const float*)d_in[p++];
  S4Params Ls[3];
  for (int li = 0; li < 3; ++li) {
    Ls[li].loglam = (const float*)d_in[p++]; Ls[li].Bm  = (const float*)d_in[p++];
    Ls[li].Cm     = (const float*)d_in[p++]; Ls[li].D   = (const float*)d_in[p++];
    Ls[li].n1g    = (const float*)d_in[p++]; Ls[li].n1b = (const float*)d_in[p++];
    Ls[li].n2g    = (const float*)d_in[p++]; Ls[li].n2b = (const float*)d_in[p++];
    Ls[li].c1w    = (const float*)d_in[p++]; Ls[li].c1b = (const float*)d_in[p++];
    Ls[li].c2w    = (const float*)d_in[p++]; Ls[li].c2b = (const float*)d_in[p++];
  }

  // Workspace layout (floats), all offsets 1KB-aligned.
  float* ws = (float*)d_ws;
  float* meanMax = ws;           float* stdMax = ws + 256;
  float* meanMin = ws + 512;     float* stdMin = ws + 768;
  float* zbuf = ws + 1024;                       // BL*DV      = 262144
  float* cbuf = zbuf + BL_ * DV_;                // BL*EH      = 2097152
  float* bufA = cbuf + (size_t)BL_ * EH_;        // BL*DM      = 4194304
  float* bufB = bufA + (size_t)BL_ * DM_;        // BL*DM
  float* bufC = bufB + (size_t)BL_ * DM_;        // BL*DM
  float* bufF = bufC + (size_t)BL_ * DM_;        // BL*DF      = 8388608
  float* ubuf = bufF;                            // overlap: dead before FFN use
  __bf16* wbf = (__bf16*)(bufF + (size_t)BL_ * DF_);  // up to DM*DF bf16

  // 1) moments
  stats_kernel<<<B_ * DV_, 256, 0, stream>>>(max_idx, min_idx, meanMax, stdMax, meanMin, stdMin);
  // 2) gating/blend -> z
  gate_kernel<<<BL_ / 8, 256, 0, stream>>>(seq_x, mask, max_idx, min_idx, max_val, min_val,
      meanMax, stdMax, meanMin, stdMin, d1w, d1b, d2w, d2b, zbuf);
  // 3) conv + mask projection -> c, u = c*m
  conv_mask_kernel<<<BL_, 256, 0, stream>>>(zbuf, mask, mem_w, mem_b, mask_w, mask_b, cbuf, ubuf);
  // 4) S4 layer 0 (H=EH): scan(u)+c, LN, FFN(WMMA), LN -> bufA
  run_s4(stream, Ls[0], ubuf, cbuf, EH_, bufC, bufB, bufF, bufC, wbf, bufA);
  // 5) lin1: [BL,256] -> [BL,512]
  cvt_bf16_kernel<<<(DM_ * EH_ + 255) / 256, 256, 0, stream>>>(lin1_w, wbf, DM_ * EH_);
  launch_gemm(stream, bufA, wbf, lin1_b, nullptr, bufC, BL_, DM_, EH_, 0);
  // 6) S4 layers 1,2 (H=DM), ungated (U == Xin)
  run_s4(stream, Ls[1], bufC, bufC, DM_, bufA, bufB, bufF, bufA, wbf, bufC);
  run_s4(stream, Ls[2], bufC, bufC, DM_, bufA, bufB, bufF, bufA, wbf, bufC);
  // 7) final LN + output projection [BL,512] -> [BL,32]
  ln_kernel<<<BL_, 256, 0, stream>>>(bufC, norm_g, norm_b, bufB, DM_);
  cvt_bf16_kernel<<<(DV_ * DM_ + 255) / 256, 256, 0, stream>>>(out_w, wbf, DV_ * DM_);
  launch_gemm(stream, bufB, wbf, out_b, nullptr, (float*)d_out, BL_, DV_, DM_, 0);
}
